// PoolNetCompressor_36180804501538
// MI455X (gfx1250) — compile-verified, matrix operand order checked by reading, not run on hardware
//
#include <hip/hip_runtime.h>
#include <hip/hip_bf16.h>
#include <cstdint>

// ---------------------------------------------------------------------------
// Types for CDNA5 WMMA / vector memory
// ---------------------------------------------------------------------------
typedef __attribute__((ext_vector_type(16))) __bf16 v16bf;
typedef __attribute__((ext_vector_type(8)))  float  v8f;
typedef __attribute__((ext_vector_type(4)))  float  v4f;
typedef __attribute__((ext_vector_type(4)))  int    v4i;

union ABf {
    v16bf        v;
    __bf16       h[16];
    unsigned int u[8];
};

// Native f32 -> bf16 convert (backend pairs these into v_cvt_pk_bf16_f32).
__device__ __forceinline__ unsigned int pack2bf(float a, float b) {
    union { __bf16 h[2]; unsigned int u; } p;
    p.h[0] = (__bf16)a;
    p.h[1] = (__bf16)b;
    return p.u;
}

// ---------------------------------------------------------------------------
// CDNA5 async global->LDS copy (ASYNCcnt-tracked), guarded so the file still
// compiles if this toolchain lacks the builtins (falls back to sync copy).
// Probe-derived signature: (v4i addrspace(1)*, v4i addrspace(3)*, imm, imm).
// ---------------------------------------------------------------------------
#if defined(__AMDGCN__) && \
    __has_builtin(__builtin_amdgcn_global_load_async_to_lds_b128) && \
    __has_builtin(__builtin_amdgcn_s_wait_asynccnt)
#define HAVE_ASYNC_LDS 1
#else
#define HAVE_ASYNC_LDS 0
#endif

#if HAVE_ASYNC_LDS
typedef __attribute__((address_space(1))) v4i* gptr_v4i;
typedef __attribute__((address_space(3))) v4i* lptr_v4i;
__device__ __forceinline__ void async_cp_b128(const uint4* gsrc, uint4* lds_dst) {
    __builtin_amdgcn_global_load_async_to_lds_b128(
        (gptr_v4i)gsrc, (lptr_v4i)lds_dst, 0, 0);
}
#endif

// ---------------------------------------------------------------------------
// Kernel 0: pack Wa1 [1024,256] f32 row-major -> bf16 B-fragments.
// Layout: packed[(chunk*16 + tile)*32 + lane][j], j=0..7 dwords.
// 16-bit B 32x16 layout: lane holds col n = 16*tile + lane%16;
//   koff = lane<16 ? 0 : 8;  dword j(0..3): K = koff+2j,+1 ; j(4..7): K += 16.
// ---------------------------------------------------------------------------
__global__ void __launch_bounds__(256) pack_wa1_kernel(
    const float* __restrict__ Wa1, unsigned int* __restrict__ packed)
{
    int tid  = blockIdx.x * 256 + threadIdx.x;    // 0..16383
    int lane = tid & 31;
    int tile = (tid >> 5) & 15;
    int c    = tid >> 9;                           // k-chunk 0..31
    int n    = tile * 16 + (lane & 15);
    int koff = (lane < 16) ? 0 : 8;
    unsigned int out[8];
#pragma unroll
    for (int j = 0; j < 8; j++) {
        int k = c * 32 + koff + ((j >= 4) ? 16 : 0) + 2 * (j & 3);
        out[j] = pack2bf(Wa1[k * 256 + n], Wa1[(k + 1) * 256 + n]);
    }
    unsigned int* dst = packed + ((size_t)tid) * 8;
#pragma unroll
    for (int j = 0; j < 8; j++) dst[j] = out[j];
}

// ---------------------------------------------------------------------------
// Kernel 1: scores[b*4096+s] = tanh(x @ Wa1 + ba1) @ Wa2 + ba2
// Grid: 2048 WGs x 256 thr (8 waves). Wave w -> tokens [wg*128+16w, +16).
// K-outer loop (x streamed once, NT), 16 N-tiles resident in accumulators.
// Wa1 chunks double-buffered in LDS via async global->LDS copies.
// ---------------------------------------------------------------------------
__global__ void __launch_bounds__(256) scores_kernel(
    const float* __restrict__ x, const unsigned int* __restrict__ wa1p,
    const float* __restrict__ ba1, const float* __restrict__ Wa2,
    const float* __restrict__ ba2, float* __restrict__ scores)
{
    __shared__ __align__(16) uint4 lds[2][1024];   // 2 x 16 KB Wa1 chunks
    const int wave = threadIdx.x >> 5;
    const int lane = threadIdx.x & 31;
    const long tokenBase = (long)blockIdx.x * 128 + wave * 16;
    const int  row  = lane & 15;
    const int  koff = (lane < 16) ? 0 : 8;
    const float* xrow = x + (tokenBase + row) * 1024;
    const uint4* wsrc = (const uint4*)wa1p;

    v8f acc[16];
#pragma unroll
    for (int t = 0; t < 16; t++)
#pragma unroll
        for (int i = 0; i < 8; i++) acc[t][i] = 0.0f;

    // prologue: stage chunk 0 into buffer 0 (4 async b128 per thread)
    for (int i = threadIdx.x; i < 1024; i += 256) {
#if HAVE_ASYNC_LDS
        async_cp_b128(wsrc + i, &lds[0][i]);
#else
        lds[0][i] = wsrc[i];
#endif
    }

    for (int c = 0; c < 32; c++) {
        // prefetch next chunk into the other buffer while we compute this one
        if (c + 1 < 32) {
            const uint4* nsrc = wsrc + (size_t)(c + 1) * 1024;
            uint4* nb = lds[(c + 1) & 1];
            for (int i = threadIdx.x; i < 1024; i += 256) {
#if HAVE_ASYNC_LDS
                async_cp_b128(nsrc + i, &nb[i]);
#else
                nb[i] = nsrc[i];
#endif
            }
#if HAVE_ASYNC_LDS
            __builtin_amdgcn_s_wait_asynccnt(4);   // chunk c's 4 ops retired
#endif
        } else {
#if HAVE_ASYNC_LDS
            __builtin_amdgcn_s_wait_asynccnt(0);
#endif
        }
        __syncthreads();                            // chunk c visible WG-wide

        // A fragment: x[row, c*32+koff .. +7] and [.. +16 .. +23] (NT, cvt bf16)
        ABf a;
        const float* p0 = xrow + c * 32 + koff;
        v4f f0 = __builtin_nontemporal_load((const v4f*)(p0));
        v4f f1 = __builtin_nontemporal_load((const v4f*)(p0 + 4));
        v4f f2 = __builtin_nontemporal_load((const v4f*)(p0 + 16));
        v4f f3 = __builtin_nontemporal_load((const v4f*)(p0 + 20));
        a.h[0]  = (__bf16)f0.x; a.h[1]  = (__bf16)f0.y;
        a.h[2]  = (__bf16)f0.z; a.h[3]  = (__bf16)f0.w;
        a.h[4]  = (__bf16)f1.x; a.h[5]  = (__bf16)f1.y;
        a.h[6]  = (__bf16)f1.z; a.h[7]  = (__bf16)f1.w;
        a.h[8]  = (__bf16)f2.x; a.h[9]  = (__bf16)f2.y;
        a.h[10] = (__bf16)f2.z; a.h[11] = (__bf16)f2.w;
        a.h[12] = (__bf16)f3.x; a.h[13] = (__bf16)f3.y;
        a.h[14] = (__bf16)f3.z; a.h[15] = (__bf16)f3.w;

        const uint4* cb = lds[c & 1];
#pragma unroll
        for (int t = 0; t < 16; t++) {
            ABf bfrag;
            const uint4* bp = cb + (t * 32 + lane) * 2;
            *(uint4*)&bfrag.u[0] = bp[0];
            *(uint4*)&bfrag.u[4] = bp[1];
            acc[t] = __builtin_amdgcn_wmma_f32_16x16x32_bf16(
                false, a.v, false, bfrag.v, (short)0, acc[t], false, false);
        }
        __syncthreads();    // compute(c) done; iter c+1 may overwrite buf[c&1]
    }

    // Epilogue: +ba1, tanh, *Wa2, reduce over N (16 tiles x 16 lanes)
    float sc[8];
#pragma unroll
    for (int r = 0; r < 8; r++) sc[r] = 0.0f;
    const int nbase = lane & 15;
#pragma unroll
    for (int t = 0; t < 16; t++) {
        int n = t * 16 + nbase;
        float bb = ba1[n];
        float w2 = Wa2[n];
#pragma unroll
        for (int r = 0; r < 8; r++)
            sc[r] += tanhf(acc[t][r] + bb) * w2;
    }
#pragma unroll
    for (int off = 1; off < 16; off <<= 1)
#pragma unroll
        for (int r = 0; r < 8; r++)
            sc[r] += __shfl_xor(sc[r], off, 32);

    if ((lane & 15) == 0) {
        long mbase = tokenBase + ((lane >= 16) ? 8 : 0);
        float b2 = ba2[0];
#pragma unroll
        for (int r = 0; r < 8; r++) scores[mbase + r] = sc[r] + b2;
    }
}

// ---------------------------------------------------------------------------
// Kernel 2: exact top-K (K=2048) via 32-bit radix bisection on monotone keys,
// tie-break by index, then mean-pool selected rows of x. One WG per batch.
// ---------------------------------------------------------------------------
__global__ void __launch_bounds__(256) topk_pool_kernel(
    const float* __restrict__ x, const float* __restrict__ scores,
    float* __restrict__ pooled)
{
    __shared__ unsigned int keys[4096];
    __shared__ int s_cnt;
    __shared__ int scanbuf[256];
    __shared__ unsigned char flag[4096];
    const int b = blockIdx.x;
    const int t = threadIdx.x;
    const int K = 2048;
    const float* sb = scores + b * 4096;

    for (int i = t; i < 4096; i += 256) {
        unsigned int u = __float_as_uint(sb[i]);
        keys[i] = (u & 0x80000000u) ? ~u : (u | 0x80000000u);
    }
    __syncthreads();

    unsigned int T = 0;
    for (int bit = 31; bit >= 0; bit--) {
        unsigned int cand = T | (1u << bit);
        if (t == 0) s_cnt = 0;
        __syncthreads();
        int local = 0;
        for (int i = t; i < 4096; i += 256) local += (keys[i] >= cand);
        atomicAdd(&s_cnt, local);
        __syncthreads();
        if (s_cnt >= K) T = cand;       // Kth-largest key, bit by bit
        __syncthreads();
    }

    // per-thread contiguous chunk: counts of >T and ==T
    const int base = t * 16;
    int gt = 0, eq = 0;
#pragma unroll
    for (int i = 0; i < 16; i++) {
        unsigned int k = keys[base + i];
        gt += (k > T); eq += (k == T);
    }
    if (t == 0) s_cnt = 0;
    __syncthreads();
    atomicAdd(&s_cnt, gt);
    __syncthreads();
    const int need = K - s_cnt;         // ties to admit, in index order
    __syncthreads();
    scanbuf[t] = eq;
    __syncthreads();
    for (int off = 1; off < 256; off <<= 1) {   // Hillis-Steele inclusive scan
        int v = scanbuf[t];
        int add = (t >= off) ? scanbuf[t - off] : 0;
        __syncthreads();
        scanbuf[t] = v + add;
        __syncthreads();
    }
    int rank = scanbuf[t] - eq;         // exclusive prefix of eq
#pragma unroll
    for (int i = 0; i < 16; i++) {
        unsigned int k = keys[base + i];
        int sel = (k > T);
        if (k == T) { sel |= (rank < need); rank++; }
        flag[base + i] = (unsigned char)sel;
    }
    __syncthreads();

    // mean pool selected rows; thread owns 4 contiguous columns (coalesced)
    float a0 = 0.f, a1 = 0.f, a2 = 0.f, a3 = 0.f;
    const float* xb = x + (long)b * 4096 * 1024;
    const int col = t * 4;
    for (int s = 0; s < 4096; s++) {
        if (flag[s]) {
            v4f v = __builtin_nontemporal_load(
                (const v4f*)(xb + (long)s * 1024 + col));
            a0 += v.x; a1 += v.y; a2 += v.z; a3 += v.w;
        }
    }
    const float inv = 1.0f / 2048.0f;
    float* pb = pooled + b * 1024 + col;
    pb[0] = a0 * inv; pb[1] = a1 * inv; pb[2] = a2 * inv; pb[3] = a3 * inv;
}

// ---------------------------------------------------------------------------
// Kernel 3a: h = GELU(LayerNorm(pooled @ Wn1 + bn1) * g + b). One WG per row.
// ---------------------------------------------------------------------------
__global__ void __launch_bounds__(256) mlp1_kernel(
    const float* __restrict__ pooled, const float* __restrict__ Wn1,
    const float* __restrict__ bn1, const float* __restrict__ g,
    const float* __restrict__ beta, float* __restrict__ hact)
{
    __shared__ float sp[1024];
    __shared__ float red[256];
    const int b = blockIdx.x, t = threadIdx.x;
    for (int i = t; i < 1024; i += 256) sp[i] = pooled[b * 1024 + i];
    __syncthreads();
    float h0 = bn1[t], h1 = bn1[t + 256];
    for (int k = 0; k < 1024; k++) {
        float p = sp[k];
        h0 += p * Wn1[k * 512 + t];
        h1 += p * Wn1[k * 512 + t + 256];
    }
    red[t] = h0 + h1;
    __syncthreads();
    for (int off = 128; off > 0; off >>= 1) { if (t < off) red[t] += red[t + off]; __syncthreads(); }
    const float mu = red[0] * (1.0f / 512.0f);
    __syncthreads();
    red[t] = (h0 - mu) * (h0 - mu) + (h1 - mu) * (h1 - mu);
    __syncthreads();
    for (int off = 128; off > 0; off >>= 1) { if (t < off) red[t] += red[t + off]; __syncthreads(); }
    const float rs = rsqrtf(red[0] * (1.0f / 512.0f) + 1e-5f);
    float n0 = (h0 - mu) * rs * g[t] + beta[t];
    float n1 = (h1 - mu) * rs * g[t + 256] + beta[t + 256];
    const float is2 = 0.70710678118654752f;
    hact[b * 512 + t]       = 0.5f * n0 * (1.0f + erff(n0 * is2));
    hact[b * 512 + t + 256] = 0.5f * n1 * (1.0f + erff(n1 * is2));
}

// ---------------------------------------------------------------------------
// Kernel 3b: out = h @ Wn2 + bn2 + pooled @ Wr + br. One WG per row.
// ---------------------------------------------------------------------------
__global__ void __launch_bounds__(256) mlp2_kernel(
    const float* __restrict__ hact, const float* __restrict__ Wn2,
    const float* __restrict__ bn2, const float* __restrict__ pooled,
    const float* __restrict__ Wr, const float* __restrict__ br,
    float* __restrict__ out)
{
    __shared__ float sh[512];
    __shared__ float sp[1024];
    const int b = blockIdx.x, t = threadIdx.x;
    for (int i = t; i < 512; i += 256)  sh[i] = hact[b * 512 + i];
    for (int i = t; i < 1024; i += 256) sp[i] = pooled[b * 1024 + i];
    __syncthreads();
    float acc = bn2[t] + br[t];
    for (int j = 0; j < 512; j++)  acc += sh[j] * Wn2[j * 256 + t];
    for (int k = 0; k < 1024; k++) acc += sp[k] * Wr[k * 256 + t];
    out[b * 256 + t] = acc;
}

// ---------------------------------------------------------------------------
// Launch
// ---------------------------------------------------------------------------
extern "C" void kernel_launch(void* const* d_in, const int* in_sizes, int n_in,
                              void* d_out, int out_size, void* d_ws, size_t ws_size,
                              hipStream_t stream) {
    const float* x   = (const float*)d_in[0];
    const float* Wa1 = (const float*)d_in[1];
    const float* ba1 = (const float*)d_in[2];
    const float* Wa2 = (const float*)d_in[3];
    const float* ba2 = (const float*)d_in[4];
    const float* Wn1 = (const float*)d_in[5];
    const float* bn1 = (const float*)d_in[6];
    const float* g   = (const float*)d_in[7];
    const float* bb  = (const float*)d_in[8];
    const float* Wn2 = (const float*)d_in[9];
    const float* bn2 = (const float*)d_in[10];
    const float* Wr  = (const float*)d_in[11];
    const float* br  = (const float*)d_in[12];

    char* ws = (char*)d_ws;
    unsigned int* wa1p = (unsigned int*)ws;                       // 512 KB
    float* scores = (float*)(ws + (512u << 10));                  // 1 MB
    float* pooled = (float*)(ws + (512u << 10) + (1024u << 10));  // 256 KB
    float* hact   = (float*)(ws + (512u << 10) + (1024u << 10) + (256u << 10)); // 128 KB

    pack_wa1_kernel <<<64,   256, 0, stream>>>(Wa1, wa1p);
    scores_kernel   <<<2048, 256, 0, stream>>>(x, wa1p, ba1, Wa2, ba2, scores);
    topk_pool_kernel<<<64,   256, 0, stream>>>(x, scores, pooled);
    mlp1_kernel     <<<64,   256, 0, stream>>>(pooled, Wn1, bn1, g, bb, hact);
    mlp2_kernel     <<<64,   256, 0, stream>>>(hact, Wn2, bn2, pooled, Wr, br, (float*)d_out);
}
